// SingleViewAttentionLSTMBasedRegressor_55353538511278
// MI455X (gfx1250) — matrix operand order repaired and checked by reading, not compiled, not run on hardware
//
#include <hip/hip_runtime.h>

// Problem constants (from reference): G genes, T timepoints, H hidden, 2 layers
constexpr int GENES  = 384;
constexpr int TPTS   = 192;
constexpr int HID    = 384;
constexpr int KTOT   = 768;   // K = [X | H] = 384 + 384
constexpr int KSTEPS = 24;    // 768 / 32 per bf16 WMMA
constexpr int MROWS  = 400;   // 385 batch rows padded to 25 tiles of 16
constexpr int MTILES = 25;
constexpr int NTILES = 24;    // HID / 16

typedef __attribute__((ext_vector_type(16))) __bf16 v16bf;
typedef __attribute__((ext_vector_type(8)))  __bf16 v8bf;
typedef __attribute__((ext_vector_type(8)))  float  v8f;

__device__ __forceinline__ unsigned short f2bfu(float f) {
  unsigned u = __float_as_uint(f);
  u += 0x7FFFu + ((u >> 16) & 1u);          // round-to-nearest-even
  return (unsigned short)(u >> 16);
}
__device__ __forceinline__ float sigmoidf_(float x) { return 1.0f / (1.0f + expf(-x)); }

// ---- pack [W_ih | W_hh] -> bf16 (1536 x 768) per layer, fuse biases --------
__global__ void pack_weights(const float* __restrict__ Wih, const float* __restrict__ Whh,
                             const float* __restrict__ bih, const float* __restrict__ bhh,
                             unsigned short* __restrict__ Wcat, float* __restrict__ bcat)
{
  int idx = blockIdx.x * blockDim.x + threadIdx.x;
  const int total = 2 * 4 * HID * KTOT;
  if (idx >= total) return;
  int k = idx % KTOT;
  int n = (idx / KTOT) % (4 * HID);
  int l = idx / (KTOT * 4 * HID);
  float v = (k < HID) ? Wih[((size_t)l * 4 * HID + n) * HID + k]
                      : Whh[((size_t)l * 4 * HID + n) * HID + (k - HID)];
  Wcat[idx] = f2bfu(v);
  if (k == 0) bcat[l * 4 * HID + n] = bih[l * 4 * HID + n] + bhh[l * 4 * HID + n];
}

// ---- zero c (fp32) and both h double-buffers (bf16) ------------------------
__global__ void init_state(float* __restrict__ c, unsigned short* __restrict__ hbuf)
{
  int idx = blockIdx.x * blockDim.x + threadIdx.x;
  if (idx < MROWS * HID) {
    c[idx] = 0.f;
    hbuf[idx] = 0;
    hbuf[MROWS * HID + idx] = 0;
  }
}

// ---- one LSTM time step: block = 16 batch rows x 16 hidden cols (all 4 gates)
template<int LAYER>
__global__ __launch_bounds__(128)
void lstm_step(int t,
               const float* __restrict__ x,              // (G, T) fp32, layer 0 only
               const unsigned short* __restrict__ W,     // (1536, 768) bf16 layer slice
               const float* __restrict__ bc,             // (1536,) fused bias
               const unsigned short* __restrict__ hprev, // (MROWS, H) bf16
               unsigned short* __restrict__ hnext,       // (MROWS, H) bf16
               float* __restrict__ cbuf,                 // (MROWS, H) fp32 in-place
               unsigned short* __restrict__ h0seq,       // (T, MROWS, H) bf16
               const float* __restrict__ fcw,            // (H,) fp32
               float* __restrict__ ypart)                // (T, NTILES, MROWS) fp32
{
  __shared__ unsigned short As[16][776];   // 16 x 768 bf16 + 8-elem pad (bank-safe)
  __shared__ float Zs[4][16][16];          // gate accumulators exchange

  const int tid = threadIdx.x;
  const int nt  = blockIdx.x;              // hidden-column tile 0..23
  const int m0  = blockIdx.y * 16;         // batch-row base 0..384
  const int nh0 = nt * 16;

  // ---- build A tile [16 x 768] = [X_t | H_prev] in bf16 -------------------
  for (int chunk = tid; chunk < 16 * 96; chunk += 128) {
    const int r   = chunk / 96;
    const int col = (chunk % 96) * 8;
    const int b   = m0 + r;
    const uint4 z4 = {0u, 0u, 0u, 0u};
    if (col < GENES) {                     // X half of K
      if (LAYER == 0) {
        *reinterpret_cast<uint4*>(&As[r][col]) = z4;
        if (b < GENES) {                   // one-hot masked gene series
          if (b >= col && b < col + 8)
            As[r][col + (b - col)] = f2bfu(x[b * TPTS + t]);
        } else if (b == GENES) {           // full series: gather x[:, t]
          #pragma unroll
          for (int j = 0; j < 8; ++j)
            As[r][col + j] = f2bfu(x[(col + j) * TPTS + t]);
        }
      } else {                             // layer 1 input = layer-0 h at time t
        if (b <= GENES)
          *reinterpret_cast<uint4*>(&As[r][col]) =
            *reinterpret_cast<const uint4*>(&h0seq[((size_t)t * MROWS + b) * HID + col]);
        else
          *reinterpret_cast<uint4*>(&As[r][col]) = z4;
      }
    } else {                               // H half of K
      const int hc = col - GENES;
      if (b <= GENES)
        *reinterpret_cast<uint4*>(&As[r][col]) =
          *reinterpret_cast<const uint4*>(&hprev[(size_t)b * HID + hc]);
      else
        *reinterpret_cast<uint4*>(&As[r][col]) = z4;
    }
  }
  __syncthreads();

  // ---- 4 waves, one gate tile each; K=768 via v_wmma_f32_16x16x32_bf16 ----
  const int wave = tid >> 5;               // == gate index (i,f,g,o)
  const int lane = tid & 31;
  const int mA   = lane & 15;
  const int aoff = (lane < 16) ? 0 : 16;
  const unsigned short* wrow =
      W + (size_t)(wave * HID + nh0 + (lane & 15)) * KTOT + ((lane < 16) ? 0 : 16);

  v8f acc = {};
  #define WMMA_K(kk)                                                              \
    {                                                                             \
      v16bf bfrag = *reinterpret_cast<const v16bf*>(wrow + (kk) * 32);            \
      v8bf a0 = *reinterpret_cast<const v8bf*>(&As[mA][(kk) * 32 + aoff]);        \
      v8bf a1 = *reinterpret_cast<const v8bf*>(&As[mA][(kk) * 32 + aoff + 8]);    \
      v16bf afrag = __builtin_shufflevector(a0, a1, 0, 1, 2, 3, 4, 5, 6, 7,       \
                                            8, 9, 10, 11, 12, 13, 14, 15);        \
      acc = __builtin_amdgcn_wmma_f32_16x16x32_bf16(false, afrag, false, bfrag,   \
                                                    (short)0, acc, false, false); \
    }

  if (LAYER == 0 && m0 < GENES) {
    // one-hot rows: X half contributes only in K-chunk m0/32 -> 13 WMMAs not 24
    WMMA_K(m0 >> 5);
    #pragma unroll
    for (int kk = KSTEPS / 2; kk < KSTEPS; ++kk) WMMA_K(kk);
  } else {
    #pragma unroll
    for (int kk = 0; kk < KSTEPS; ++kk) WMMA_K(kk);
  }
  #undef WMMA_K

  #pragma unroll
  for (int i = 0; i < 8; ++i)
    Zs[wave][i + ((lane < 16) ? 0 : 8)][lane & 15] = acc[i];
  __syncthreads();

  // ---- gate fusion + state update: 2 elements per thread ------------------
  const int row = tid >> 3;
  const int c0  = (tid & 7) * 2;
  float hval[2];
  #pragma unroll
  for (int j = 0; j < 2; ++j) {
    const int c  = c0 + j;
    const float zi = Zs[0][row][c] + bc[0 * HID + nh0 + c];
    const float zf = Zs[1][row][c] + bc[1 * HID + nh0 + c];
    const float zg = Zs[2][row][c] + bc[2 * HID + nh0 + c];
    const float zo = Zs[3][row][c] + bc[3 * HID + nh0 + c];
    const float ig = sigmoidf_(zi);
    const float fg = sigmoidf_(zf);
    const float gg = tanhf(zg);
    const float og = sigmoidf_(zo);
    const size_t ci = (size_t)(m0 + row) * HID + nh0 + c;
    const float cn  = fg * cbuf[ci] + ig * gg;
    cbuf[ci] = cn;
    const float h = og * tanhf(cn);
    hval[j] = h;
    const unsigned short hb = f2bfu(h);
    hnext[ci] = hb;
    if (LAYER == 0)
      h0seq[((size_t)t * MROWS + m0 + row) * HID + nh0 + c] = hb;
  }

  if (LAYER == 1) {
    __syncthreads();                       // all Zs gate reads complete
    Zs[0][row][c0]     = hval[0];
    Zs[0][row][c0 + 1] = hval[1];
    __syncthreads();
    if (tid < 16) {                        // deterministic per-tile partial fc dot
      float s = 0.f;
      #pragma unroll
      for (int cc = 0; cc < 16; ++cc)
        s += Zs[0][tid][cc] * fcw[nh0 + cc];
      ypart[((size_t)t * NTILES + nt) * MROWS + m0 + tid] = s;
    }
  }
}

// ---- reduce fc partials -> y (385 x 192), emit yi_pred --------------------
__global__ void finalize_y(const float* __restrict__ ypart, const float* __restrict__ fcb,
                           float* __restrict__ y, float* __restrict__ out_yi)
{
  int idx = blockIdx.x * blockDim.x + threadIdx.x;
  if (idx >= (GENES + 1) * TPTS) return;
  int b = idx / TPTS, t = idx % TPTS;
  float s = fcb[0];
  #pragma unroll
  for (int nt = 0; nt < NTILES; ++nt)
    s += ypart[((size_t)t * NTILES + nt) * MROWS + b];
  y[b * TPTS + t] = s;
  if (b < GENES) out_yi[b * TPTS + t] = s;   // yi_pred region of d_out
}

// ---- softmax attention + y_pred (single block, tiny) ----------------------
__global__ void attention_kernel(const float* __restrict__ y, float* __restrict__ out)
{
  __shared__ float sc[GENES];
  __shared__ float red[2];
  const int i = threadIdx.x;                 // 0..383
  const float* yall = y + (size_t)GENES * TPTS;
  float s = 0.f;
  for (int t = 0; t < TPTS; ++t) s += yall[t] * y[i * TPTS + t];
  sc[i] = s;
  __syncthreads();
  if (i == 0) {
    float mx = sc[0];
    for (int j = 1; j < GENES; ++j) mx = fmaxf(mx, sc[j]);
    red[0] = mx;
  }
  __syncthreads();
  sc[i] = expf(sc[i] - red[0]);
  __syncthreads();
  if (i == 0) {
    float sm = 0.f;
    for (int j = 0; j < GENES; ++j) sm += sc[j];
    red[1] = sm;
  }
  __syncthreads();
  const float a = sc[i] / red[1];
  out[TPTS + GENES * TPTS + i] = a;          // attn
  sc[i] = a;
  __syncthreads();
  if (i < TPTS) {
    float yp = 0.f;
    for (int j = 0; j < GENES; ++j) yp += sc[j] * y[j * TPTS + i];
    out[i] = yp;                             // y_pred
  }
}

extern "C" void kernel_launch(void* const* d_in, const int* in_sizes, int n_in,
                              void* d_out, int out_size, void* d_ws, size_t ws_size,
                              hipStream_t stream)
{
  (void)in_sizes; (void)n_in; (void)out_size; (void)ws_size;
  const float* x   = (const float*)d_in[0];
  const float* Wih = (const float*)d_in[1];
  const float* Whh = (const float*)d_in[2];
  const float* bih = (const float*)d_in[3];
  const float* bhh = (const float*)d_in[4];
  const float* fcw = (const float*)d_in[5];
  const float* fcb = (const float*)d_in[6];
  float* out = (float*)d_out;

  char* ws = (char*)d_ws;
  size_t off = 0;
  unsigned short* Wcat = (unsigned short*)(ws + off); off += (size_t)2 * 1536 * KTOT * 2;
  float* bcat          = (float*)(ws + off);          off += (size_t)2 * 1536 * 4;
  float* cbuf          = (float*)(ws + off);          off += (size_t)MROWS * HID * 4;
  unsigned short* hbuf = (unsigned short*)(ws + off); off += (size_t)2 * MROWS * HID * 2;
  unsigned short* h0sq = (unsigned short*)(ws + off); off += (size_t)TPTS * MROWS * HID * 2;
  float* ypart         = (float*)(ws + off);          off += (size_t)TPTS * NTILES * MROWS * 4;
  float* ybuf          = (float*)(ws + off);          off += (size_t)(GENES + 1) * TPTS * 4;

  pack_weights<<<(2 * 1536 * KTOT + 255) / 256, 256, 0, stream>>>(Wih, Whh, bih, bhh, Wcat, bcat);

  const dim3 grid(NTILES, MTILES);
  const int stbl = (MROWS * HID + 255) / 256;
  const int HS = MROWS * HID;

  init_state<<<stbl, 256, 0, stream>>>(cbuf, hbuf);
  for (int t = 0; t < TPTS; ++t)
    lstm_step<0><<<grid, 128, 0, stream>>>(t, x, Wcat, bcat,
        hbuf + (t & 1) * HS, hbuf + ((t + 1) & 1) * HS, cbuf, h0sq, fcw, ypart);

  init_state<<<stbl, 256, 0, stream>>>(cbuf, hbuf);
  for (int t = 0; t < TPTS; ++t)
    lstm_step<1><<<grid, 128, 0, stream>>>(t, x, Wcat + (size_t)1536 * KTOT, bcat + 1536,
        hbuf + (t & 1) * HS, hbuf + ((t + 1) & 1) * HS, cbuf, h0sq, fcw, ypart);

  finalize_y<<<((GENES + 1) * TPTS + 255) / 256, 256, 0, stream>>>(ypart, fcb, ybuf, out + TPTS);
  attention_kernel<<<1, GENES, 0, stream>>>(ybuf, out);
}
